// GATGNN_47974784696364
// MI455X (gfx1250) — matrix operand order repaired and checked by reading, not compiled, not run on hardware
//
#include <hip/hip_runtime.h>
#include <hip/hip_bf16.h>

#define NN 30000
#define EE 480000
#define GG 128
#define HH 4
#define DD 64
#define NFD 92
#define EFD 50
#define GFD 108
#define LLAYERS 5

typedef __attribute__((ext_vector_type(16))) __bf16 v16bf;
typedef __attribute__((ext_vector_type(8)))  float  v8f;

static __device__ __forceinline__ unsigned short f2bfbits(float f) {
  unsigned u = __builtin_bit_cast(unsigned, f);
  unsigned r = (u + 0x7FFFu + ((u >> 16) & 1u)) >> 16;   // round-to-nearest-even
  return (unsigned short)r;
}
static __device__ __forceinline__ float lrelu(float t) { return t >= 0.f ? t : 0.2f * t; }

// ---------------------------------------------------------------- zero
__global__ void k_zero(float* p, int n) {
  int i = blockIdx.x * 256 + threadIdx.x;
  if (i < n) p[i] = 0.f;
}

// ---------------------------------------------------------------- embeddings
__global__ void k_embed_nodes(const float* __restrict__ x, const float* __restrict__ W,
                              const float* __restrict__ b, float* __restrict__ xemb) {
  int i = blockIdx.x * 256 + threadIdx.x;
  if (i >= NN * DD) return;
  int n = i >> 6, d = i & 63;
  const float* xp = x + (size_t)n * NFD;
  float s = b[d];
  for (int k = 0; k < NFD; ++k) s += xp[k] * W[k * DD + d];
  xemb[i] = lrelu(s);
}

__global__ void k_embed_edges(const float* __restrict__ ea, const float* __restrict__ W,
                              const float* __restrict__ b, float* __restrict__ eaemb) {
  size_t i = (size_t)blockIdx.x * 256 + threadIdx.x;
  if (i >= (size_t)EE * DD) return;
  size_t e = i >> 6; int d = (int)(i & 63);
  const float* ep = ea + e * EFD;
  float s = b[d];
  for (int k = 0; k < EFD; ++k) s += ep[k] * W[k * DD + d];
  eaemb[i] = lrelu(s);
}

// ---------------------------------------------------------------- weight prep
// Pack conv_W[l] (128x256 f32) into bf16 WMMA B-fragment order:
// wfrag[l][ks][nt][lane][i], element i of lane = W[ks*32 + (lane>>4)*16 + i][nt*16 + (lane&15)]
__global__ void k_prep_wfrag(const float* __restrict__ convW, unsigned short* __restrict__ wf) {
  int tid = blockIdx.x * 256 + threadIdx.x;
  if (tid >= LLAYERS * 4 * 16 * 32) return;
  int lane = tid & 31;
  int nt   = (tid >> 5) & 15;
  int ks   = (tid >> 9) & 3;
  int l    = tid >> 11;
  int n    = nt * 16 + (lane & 15);
  int kb   = ks * 32 + (lane >> 4) * 16;
  unsigned short* out = wf + (size_t)tid * 16;
  const float* Wl = convW + (size_t)l * 128 * 256;
#pragma unroll
  for (int i = 0; i < 16; ++i)
    out[i] = f2bfbits(Wl[(kb + i) * 256 + n]);
}

// Wa_i[l][k][h] = sum_d W[l][k][h*64+d]*att[l][h][d] ; Wa_j uses att[...][64+d]
__global__ void k_prep_wa(const float* __restrict__ convW, const float* __restrict__ att,
                          float* __restrict__ WaI, float* __restrict__ WaJ) {
  int tid = blockIdx.x * 256 + threadIdx.x;
  if (tid >= LLAYERS * 128 * 4) return;
  int h = tid & 3;
  int k = (tid >> 2) & 127;
  int l = tid >> 9;
  const float* Wl = convW + ((size_t)l * 128 + k) * 256 + h * 64;
  const float* al = att + ((size_t)l * 4 + h) * 128;
  float si = 0.f, sj = 0.f;
  for (int d = 0; d < 64; ++d) { si += Wl[d] * al[d]; sj += Wl[d] * al[64 + d]; }
  WaI[((size_t)l * 128 + k) * 4 + h] = si;
  WaJ[((size_t)l * 128 + k) * 4 + h] = sj;
}

// ---------------------------------------------------------------- pass A: scores
__global__ void __launch_bounds__(256)
k_scores(const float* __restrict__ xemb, const float* __restrict__ eaemb,
         const float* __restrict__ WaI, const float* __restrict__ WaJ,
         const float* __restrict__ bn_g, const float* __restrict__ bn_b,
         const float* __restrict__ bn_m, const float* __restrict__ bn_v,
         const int* __restrict__ rowI, const int* __restrict__ colI,
         float* __restrict__ a_un, float* __restrict__ normbuf) {
  __shared__ float sI[512], sJ[512];
  for (int i = threadIdx.x; i < 512; i += 256) { sI[i] = WaI[i]; sJ[i] = WaJ[i]; }
  __syncthreads();
  int e = blockIdx.x * 256 + threadIdx.x;
  if (e >= EE) return;
  int r = rowI[e], c = colI[e];
  const float* xr = xemb + (size_t)r * DD;
  const float* xc = xemb + (size_t)c * DD;
  const float* ep = eaemb + (size_t)e * DD;
  float si[4] = {0, 0, 0, 0}, sj[4] = {0, 0, 0, 0};
  for (int k = 0; k < 64; ++k) {
    float a = xr[k], b = xc[k], g = ep[k];
#pragma unroll
    for (int h = 0; h < 4; ++h) {
      si[h] += a * sI[k * 4 + h] + g * sI[(64 + k) * 4 + h];
      sj[h] += b * sJ[k * 4 + h] + g * sJ[(64 + k) * 4 + h];
    }
  }
#pragma unroll
  for (int h = 0; h < 4; ++h) {
    float s  = lrelu(si[h] + sj[h]);
    float bn = (s - bn_m[h]) * rsqrtf(bn_v[h] + 1e-3f) * bn_g[h] + bn_b[h];
    float a  = __expf(bn);
    a_un[(size_t)e * 4 + h] = a;
    atomicAdd(&normbuf[(size_t)r * 4 + h], a);
  }
}

// ---------------------------------------------------------------- pass A2: normalize weights in place
// a_un[e,h] <- a_un[e,h] / norm[row[e],h]   (edge softmax; one divide per (e,h))
__global__ void k_wnorm(float* __restrict__ a_un, const float* __restrict__ normbuf,
                        const int* __restrict__ rowI) {
  int i = blockIdx.x * 256 + threadIdx.x;
  if (i >= EE * HH) return;
  int e = i >> 2, h = i & 3;
  a_un[i] = a_un[i] / normbuf[(size_t)rowI[e] * 4 + h];
}

// ---------------------------------------------------------------- pass B: xj GEMM (WMMA) + scatter
// Block = 256 threads (8 waves), 16 edges per block, xj = concat(x[col],ea) @ W  ([16x128]@[128x256] bf16)
__global__ void __launch_bounds__(256)
k_messages(const float* __restrict__ xemb, const float* __restrict__ eaemb,
           const unsigned short* __restrict__ wfrag,   // layer base: [4][16][32][16] bf16
           const float* __restrict__ wgt,              // normalized softmax weights [E][4]
           const int* __restrict__ rowI, const int* __restrict__ colI,
           float* __restrict__ total) {
  __shared__ unsigned short sA[16][128];
  const int e0 = blockIdx.x * 16;
  const int t  = threadIdx.x;

  // stage A tile (bf16): row m = edge, k<64 -> x[col[e]], k>=64 -> ea[e]
  for (int i = t; i < 16 * 128; i += 256) {
    int m = i >> 7, k = i & 127;
    int e = e0 + m;
    float f = (k < 64) ? xemb[(size_t)colI[e] * DD + k]
                       : eaemb[(size_t)e * DD + (k - 64)];
    sA[m][k] = f2bfbits(f);
  }
  __syncthreads();

  const int lane = t & 31, wave = t >> 5;
  const int rowm = lane & 15;            // A: M = lane&15
  const int kb   = (lane >> 4) * 16;     // A: K half-select
  const int nt0  = wave * 2, nt1 = wave * 2 + 1;
  v8f acc0 = {}, acc1 = {};

#pragma unroll
  for (int ks = 0; ks < 4; ++ks) {
    v16bf af;
    int k0 = ks * 32 + kb;
#pragma unroll
    for (int i = 0; i < 8; ++i) {
      af[i]     = __builtin_bit_cast(__bf16, sA[rowm][k0 + i]);
      af[8 + i] = __builtin_bit_cast(__bf16, sA[rowm][k0 + 8 + i]);
    }
    const v16bf* bp0 = (const v16bf*)(wfrag + (((size_t)(ks * 16 + nt0) * 32 + lane) * 16));
    const v16bf* bp1 = (const v16bf*)(wfrag + (((size_t)(ks * 16 + nt1) * 32 + lane) * 16));
    acc0 = __builtin_amdgcn_wmma_f32_16x16x32_bf16(false, af, false, *bp0, (short)0, acc0, false, false);
    acc1 = __builtin_amdgcn_wmma_f32_16x16x32_bf16(false, af, false, *bp1, (short)0, acc1, false, false);
  }

  // scatter: C/D layout -> M = v + 8*(lane>>4), N = lane&15
  const int mhi = (lane >> 4) * 8;
  const int nlo = lane & 15;
#pragma unroll
  for (int v = 0; v < 8; ++v) {
    int e = e0 + v + mhi;
    {
      int c = nt0 * 16 + nlo, h = c >> 6, d = c & 63;
      float w = wgt[(size_t)e * 4 + h];
      int node = rowI[(h * EE + e) >> 2];           // faithful transpose/reshape scramble
      atomicAdd(&total[(size_t)node * DD + d], acc0[v] * w);
    }
    {
      int c = nt1 * 16 + nlo, h = c >> 6, d = c & 63;
      float w = wgt[(size_t)e * 4 + h];
      int node = rowI[(h * EE + e) >> 2];
      atomicAdd(&total[(size_t)node * DD + d], acc1[v] * w);
    }
  }
}

// ---------------------------------------------------------------- pass C: residual update
__global__ void k_update(float* __restrict__ xemb, const float* __restrict__ total,
                         const float* __restrict__ bias) {
  int i = blockIdx.x * 256 + threadIdx.x;
  if (i >= NN * DD) return;
  xemb[i] += total[i] * 0.25f + bias[i & 63];   // mean over H=4 heads + conv bias
}

// ---------------------------------------------------------------- pooling head
__global__ void k_ga0(const float* __restrict__ xemb, const float* __restrict__ gf,
                      const int* __restrict__ batch, const float* __restrict__ W0,
                      const float* __restrict__ b0, const float* __restrict__ g,
                      const float* __restrict__ b, const float* __restrict__ m,
                      const float* __restrict__ v, float* __restrict__ h0) {
  int i = blockIdx.x * 256 + threadIdx.x;
  if (i >= NN * DD) return;
  int n = i >> 6, d = i & 63;
  const float* xp = xemb + (size_t)n * DD;
  const float* gp = gf + (size_t)batch[n] * GFD;
  float s = b0[d];
  for (int k = 0; k < 64; ++k)  s += xp[k] * W0[k * DD + d];
  for (int k = 0; k < GFD; ++k) s += gp[k] * W0[(64 + k) * DD + d];
  s = (s - m[d]) * rsqrtf(v[d] + 1e-3f) * g[d] + b[d];
  h0[i] = fmaxf(s, 0.f);
}

__global__ void k_ga1(const float* __restrict__ h0, const float* __restrict__ W1,
                      const float* __restrict__ b1, const float* __restrict__ g,
                      const float* __restrict__ b, const float* __restrict__ m,
                      const float* __restrict__ v, float* __restrict__ h1) {
  int i = blockIdx.x * 256 + threadIdx.x;
  if (i >= NN * DD) return;
  int n = i >> 6, d = i & 63;
  const float* hp = h0 + (size_t)n * DD;
  float s = b1[d];
  for (int k = 0; k < 64; ++k) s += hp[k] * W1[k * DD + d];
  s = (s - m[d]) * rsqrtf(v[d] + 1e-3f) * g[d] + b[d];
  h1[i] = fmaxf(s, 0.f);
}

__global__ void k_attn(const float* __restrict__ h1, const float* __restrict__ W2,
                       const float* __restrict__ b2, const int* __restrict__ batch,
                       float* __restrict__ attn, float* __restrict__ gnorm) {
  int n = blockIdx.x * 256 + threadIdx.x;
  if (n >= NN) return;
  const float* hp = h1 + (size_t)n * DD;
  float s = b2[0];
  for (int k = 0; k < 64; ++k) s += hp[k] * W2[k];
  float a = __expf(s);
  attn[n] = a;
  atomicAdd(&gnorm[batch[n]], a);
}

__global__ void k_pool(const float* __restrict__ xemb, const float* __restrict__ attn,
                       const float* __restrict__ gnorm, const int* __restrict__ batch,
                       float* __restrict__ pooled) {
  int i = blockIdx.x * 256 + threadIdx.x;
  if (i >= NN * DD) return;
  int n = i >> 6, d = i & 63;
  int g = batch[n];
  atomicAdd(&pooled[(size_t)g * DD + d], xemb[i] * attn[n] / gnorm[g]);
}

__global__ void k_out(const float* __restrict__ pooled, const float* __restrict__ W0,
                      const float* __restrict__ b0, const float* __restrict__ W1,
                      const float* __restrict__ b1, float* __restrict__ out) {
  int g = threadIdx.x;
  if (g >= GG) return;
  const float* pp = pooled + (size_t)g * DD;
  float res = b1[0];
  for (int d = 0; d < 64; ++d) {
    float s = b0[d];
    for (int k = 0; k < 64; ++k) s += pp[k] * W0[k * DD + d];
    res += fmaxf(s, 0.f) * W1[d];
  }
  out[g] = res;
}

// ---------------------------------------------------------------- launcher
extern "C" void kernel_launch(void* const* d_in, const int* in_sizes, int n_in,
                              void* d_out, int out_size, void* d_ws, size_t ws_size,
                              hipStream_t stream) {
  const float* x       = (const float*)d_in[0];
  const float* eattr   = (const float*)d_in[1];
  const float* gfeat   = (const float*)d_in[2];
  const float* node_W  = (const float*)d_in[3];
  const float* node_b  = (const float*)d_in[4];
  const float* edge_W  = (const float*)d_in[5];
  const float* edge_b  = (const float*)d_in[6];
  const float* conv_W  = (const float*)d_in[7];
  const float* conv_att= (const float*)d_in[8];
  const float* conv_bias=(const float*)d_in[9];
  const float* bn_g    = (const float*)d_in[10];
  const float* bn_b    = (const float*)d_in[11];
  const float* bn_m    = (const float*)d_in[12];
  const float* bn_v    = (const float*)d_in[13];
  const float* ga_W0   = (const float*)d_in[14];
  const float* ga_b0   = (const float*)d_in[15];
  const float* bn0_g   = (const float*)d_in[16];
  const float* bn0_b   = (const float*)d_in[17];
  const float* bn0_m   = (const float*)d_in[18];
  const float* bn0_v   = (const float*)d_in[19];
  const float* ga_W1   = (const float*)d_in[20];
  const float* ga_b1   = (const float*)d_in[21];
  const float* bn1_g   = (const float*)d_in[22];
  const float* bn1_b   = (const float*)d_in[23];
  const float* bn1_m   = (const float*)d_in[24];
  const float* bn1_v   = (const float*)d_in[25];
  const float* ga_W2   = (const float*)d_in[26];
  const float* ga_b2   = (const float*)d_in[27];
  const float* out_W0  = (const float*)d_in[28];
  const float* out_b0  = (const float*)d_in[29];
  const float* out_W1  = (const float*)d_in[30];
  const float* out_b1  = (const float*)d_in[31];
  const int*   eidx    = (const int*)d_in[32];
  const int*   batch   = (const int*)d_in[33];
  float* out = (float*)d_out;

  const int* rowI = eidx;         // edge_index[0]
  const int* colI = eidx + EE;    // edge_index[1]

  char* ws = (char*)d_ws;
  auto carve = [&](size_t bytes) -> char* {
    char* p = ws; ws += (bytes + 255) & ~(size_t)255; return p;
  };
  float* xemb   = (float*)carve((size_t)NN * DD * 4);
  float* eaemb  = (float*)carve((size_t)EE * DD * 4);
  unsigned short* wfrag = (unsigned short*)carve((size_t)LLAYERS * 4 * 16 * 32 * 16 * 2);
  float* WaI    = (float*)carve((size_t)LLAYERS * 128 * 4 * 4);
  float* WaJ    = (float*)carve((size_t)LLAYERS * 128 * 4 * 4);
  float* a_un   = (float*)carve((size_t)EE * 4 * 4);
  float* normbuf= (float*)carve((size_t)NN * 4 * 4);
  float* total  = (float*)carve((size_t)NN * DD * 4);
  float* h0     = (float*)carve((size_t)NN * DD * 4);
  float* h1     = (float*)carve((size_t)NN * DD * 4);
  float* attn   = (float*)carve((size_t)NN * 4);
  float* gnorm  = (float*)carve((size_t)GG * 4);
  float* pooled = (float*)carve((size_t)GG * DD * 4);

  const int ND = NN * DD;
  const size_t ED = (size_t)EE * DD;

  // embeddings + weight prep
  k_embed_nodes<<<(ND + 255) / 256, 256, 0, stream>>>(x, node_W, node_b, xemb);
  k_embed_edges<<<(int)((ED + 255) / 256), 256, 0, stream>>>(eattr, edge_W, edge_b, eaemb);
  k_prep_wfrag<<<(LLAYERS * 4 * 16 * 32 + 255) / 256, 256, 0, stream>>>(conv_W, wfrag);
  k_prep_wa<<<(LLAYERS * 128 * 4 + 255) / 256, 256, 0, stream>>>(conv_W, conv_att, WaI, WaJ);

  for (int l = 0; l < LLAYERS; ++l) {
    k_zero<<<(NN * 4 + 255) / 256, 256, 0, stream>>>(normbuf, NN * 4);
    k_zero<<<(ND + 255) / 256, 256, 0, stream>>>(total, ND);
    k_scores<<<(EE + 255) / 256, 256, 0, stream>>>(
        xemb, eaemb, WaI + (size_t)l * 512, WaJ + (size_t)l * 512,
        bn_g + l * 4, bn_b + l * 4, bn_m + l * 4, bn_v + l * 4,
        rowI, colI, a_un, normbuf);
    k_wnorm<<<(EE * HH + 255) / 256, 256, 0, stream>>>(a_un, normbuf, rowI);
    k_messages<<<EE / 16, 256, 0, stream>>>(
        xemb, eaemb, wfrag + (size_t)l * 4 * 16 * 32 * 16,
        a_un, rowI, colI, total);
    k_update<<<(ND + 255) / 256, 256, 0, stream>>>(xemb, total, conv_bias + l * 64);
  }

  // global attention pooling + output MLP
  k_zero<<<1, 256, 0, stream>>>(gnorm, GG);
  k_zero<<<(GG * DD + 255) / 256, 256, 0, stream>>>(pooled, GG * DD);
  k_ga0<<<(ND + 255) / 256, 256, 0, stream>>>(xemb, gfeat, batch, ga_W0, ga_b0,
                                              bn0_g, bn0_b, bn0_m, bn0_v, h0);
  k_ga1<<<(ND + 255) / 256, 256, 0, stream>>>(h0, ga_W1, ga_b1,
                                              bn1_g, bn1_b, bn1_m, bn1_v, h1);
  k_attn<<<(NN + 255) / 256, 256, 0, stream>>>(h1, ga_W2, ga_b2, batch, attn, gnorm);
  k_pool<<<(ND + 255) / 256, 256, 0, stream>>>(xemb, attn, gnorm, batch, pooled);
  k_out<<<1, GG, 0, stream>>>(pooled, out_W0, out_b0, out_W1, out_b1, out);
}
